// Home_social_enocder_4681514352865
// MI455X (gfx1250) — matrix operand order repaired and checked by reading, not compiled, not run on hardware
//
#include <hip/hip_runtime.h>
#include <hip/hip_bf16.h>

// Problem constants (match reference)
#define B_  8192
#define N_  64
#define E_  128
#define H_  8
#define HD_ 16
#define ROWS_PER_BLOCK 4
#define THREADS 256

// Padded LDS strides (elements)
#define WPAD  136   // weight row: 128 bf16 + 8 pad  (272 B, 16B-aligned, bank-staggered)
#define APAD  136   // agent row:  128 bf16 + 8 pad
#define KVPAD 132   // K/V row:    128 f32  + 4 pad

typedef __attribute__((ext_vector_type(16))) __bf16 v16bf;
typedef __attribute__((ext_vector_type(8)))  __bf16 v8bf;
typedef __attribute__((ext_vector_type(8)))  float  v8f;

// LDS byte offsets (all 16B aligned)
#define OFF_W    0                      // 256*136*2 = 69632  (rows 0-127: Wk, 128-255: Wv, bf16)
#define OFF_A    69632                  // 64*136*2  = 17408
#define OFF_K    87040                  // 64*132*4  = 33792
#define OFF_V    120832                 // 33792
#define OFF_BK   154624                 // 512
#define OFF_BV   155136                 // 512
#define OFF_T    155648                 // 512
#define OFF_Q    156160                 // 512
#define OFF_ATT  156672                 // 8*64*4 = 2048
#define OFF_CTX  158720                 // 512
#define OFF_ENC  159232                 // 512
#define OFF_MSK  159744                 // 64*4 = 256
#define OFF_RED  160000                 // 32
#define SMEM_BYTES 160032

__device__ __forceinline__ float wave_sum(float x) {
#pragma unroll
  for (int o = 16; o > 0; o >>= 1) x += __shfl_xor(x, o, 32);
  return x;
}
__device__ __forceinline__ float wave_max(float x) {
#pragma unroll
  for (int o = 16; o > 0; o >>= 1) x = fmaxf(x, __shfl_xor(x, o, 32));
  return x;
}
__device__ __forceinline__ float dot16(const float* __restrict__ a,
                                       const float* __restrict__ b) {
  float s = 0.f;
#pragma unroll
  for (int i = 0; i < 16; ++i) s = fmaf(a[i], b[i], s);
  return s;
}
__device__ __forceinline__ void pack4_bf16(__bf16* dst, float4 v) {
  union { __bf16 h[4]; unsigned long long u; } p;
  p.h[0] = (__bf16)v.x; p.h[1] = (__bf16)v.y;
  p.h[2] = (__bf16)v.z; p.h[3] = (__bf16)v.w;
  *(unsigned long long*)dst = p.u;   // 8B-aligned ds_store_b64
}

__global__ void __launch_bounds__(THREADS, 2)
social_encoder_kernel(const float* __restrict__ target_emb,   // [B,E]
                      const float* __restrict__ agent_emb,    // [B,N,E]
                      const unsigned char* __restrict__ agent_mask, // [B,N] bool
                      const float* __restrict__ in_proj_w,    // [3E,E]
                      const float* __restrict__ in_proj_b,    // [3E]
                      const float* __restrict__ out_w,        // [E,E]
                      const float* __restrict__ out_b,        // [E]
                      const float* __restrict__ ln_g,         // [E]
                      const float* __restrict__ ln_b,         // [E]
                      float* __restrict__ out)                // [B,E]
{
  extern __shared__ char smem[];
  __bf16* sW   = (__bf16*)(smem + OFF_W);     // [256][WPAD]
  __bf16* sA   = (__bf16*)(smem + OFF_A);     // [64][APAD]
  float*  sK   = (float*) (smem + OFF_K);     // [64][KVPAD]
  float*  sV   = (float*) (smem + OFF_V);     // [64][KVPAD]
  float*  sBK  = (float*) (smem + OFF_BK);    // [128]
  float*  sBV  = (float*) (smem + OFF_BV);    // [128]
  float*  sT   = (float*) (smem + OFF_T);     // [128]
  float*  sQ   = (float*) (smem + OFF_Q);     // [128]
  float*  sAtt = (float*) (smem + OFF_ATT);   // [8][64]
  float*  sCtx = (float*) (smem + OFF_CTX);   // [128]
  float*  sEnc = (float*) (smem + OFF_ENC);   // [128]
  float*  sMsk = (float*) (smem + OFF_MSK);   // [64]
  float*  sRed = (float*) (smem + OFF_RED);   // [0]=mu [1]=rstd [2]=hasAgent

  const int tid  = threadIdx.x;
  const int wv   = tid >> 5;          // wave 0..7
  const int lane = tid & 31;

  // ---- one-time per block: stage Wk|Wv (bf16) and K/V biases -------------
#pragma unroll
  for (int it = 0; it < 32; ++it) {
    int flat = (it * THREADS + tid) * 4;       // 0..32767
    int row  = flat >> 7;                      // 0..255
    int col  = flat & 127;
    float4 w4 = *(const float4*)(in_proj_w + (size_t)(E_ + row) * E_ + col);
    pack4_bf16(sW + row * WPAD + col, w4);
  }
  if (tid < 128) {
    sBK[tid] = in_proj_b[E_ + tid];
    sBV[tid] = in_proj_b[2 * E_ + tid];
  }
  __syncthreads();

  const float scale = 0.25f;  // 1/sqrt(HD)

  for (int rr = 0; rr < ROWS_PER_BLOCK; ++rr) {
    const int row = blockIdx.x * ROWS_PER_BLOCK + rr;

    // ---- stage target row, agent tile (f32 -> bf16), mask ----------------
    if (tid < 128) sT[tid] = target_emb[(size_t)row * E_ + tid];
    if (tid < 64)  sMsk[tid] = agent_mask[(size_t)row * N_ + tid] ? 1.f : 0.f;
    const float* arow = agent_emb + (size_t)row * N_ * E_;
#pragma unroll
    for (int it = 0; it < 8; ++it) {
      int flat = (it * THREADS + tid) * 4;     // 0..8191
      int n    = flat >> 7;
      int col  = flat & 127;
      float4 a4 = *(const float4*)(arow + flat);
      pack4_bf16(sA + n * APAD + col, a4);
      (void)n; (void)col;
    }
    // prefetch next row's agent tile into caches (global_prefetch_b8)
    if (rr + 1 < ROWS_PER_BLOCK)
      __builtin_prefetch(agent_emb + (size_t)(row + 1) * N_ * E_ + tid * 32, 0, 0);
    __syncthreads();

    // ---- K / V projection with v_wmma_f32_16x16x32_bf16 ------------------
    // waves 0-3: K (Wk = sW rows 0-127); waves 4-7: V (Wv = sW rows 128-255)
    {
      const int     mt    = wv & 3;
      const __bf16* wbase = sW + ((wv < 4) ? 0 : 128) * WPAD;
      float*        dst   = (wv < 4) ? sK : sV;
      const float*  bias  = (wv < 4) ? sBK : sBV;

      const int mrow = mt * 16 + (lane & 15);
      const int kbA  = (lane < 16) ? 0 : 8;    // A lane-half K base
      const int kbB  = (lane < 16) ? 0 : 8;    // B lane-half K base

      v16bf afrag[4];
#pragma unroll
      for (int kc = 0; kc < 4; ++kc) {
        const __bf16* pa = sA + mrow * APAD + kc * 32 + kbA;
        v8bf lo = *(const v8bf*)pa;            // K = kb..kb+7
        v8bf hi = *(const v8bf*)(pa + 16);     // K = kb+16..kb+23
        v16bf a;
#pragma unroll
        for (int i = 0; i < 8; ++i) { a[i] = lo[i]; a[i + 8] = hi[i]; }
        afrag[kc] = a;
      }

#pragma unroll
      for (int nt = 0; nt < 8; ++nt) {
        const int ecol = nt * 16 + (lane & 15);
        const __bf16* wrow = wbase + ecol * WPAD;
        v8f c = {};
#pragma unroll
        for (int kc = 0; kc < 4; ++kc) {
          const __bf16* pb = wrow + kc * 32 + kbB;
          v8bf blo = *(const v8bf*)pb;         // K = kb..kb+7 (contiguous W row)
          v8bf bhi = *(const v8bf*)(pb + 16);  // K = kb+16..kb+23
          v16bf b;
#pragma unroll
          for (int i = 0; i < 8; ++i) { b[i] = blo[i]; b[i + 8] = bhi[i]; }
          c = __builtin_amdgcn_wmma_f32_16x16x32_bf16(
                  false, afrag[kc], false, b, (short)0, c, false, false);
        }
        const int rbase = mt * 16 + ((lane < 16) ? 0 : 8);
        const float be = bias[ecol];
#pragma unroll
        for (int i = 0; i < 8; ++i)
          dst[(rbase + i) * KVPAD + ecol] = c[i] + be;
      }
    }
    __syncthreads();

    // ---- q projection (VALU, Wq = in_proj_w rows 0..127, L2/WGP$ hot) ----
    if (tid < 128) {
      const float* wq = in_proj_w + (size_t)tid * E_;
      float acc = in_proj_b[tid];
#pragma unroll 8
      for (int j = 0; j < 128; ++j) acc = fmaf(sT[j], wq[j], acc);
      sQ[tid] = acc;
    }
    __syncthreads();

    // ---- scores + masked softmax: wave h owns head h ----------------------
    {
      const int h = wv;
      const float* qh = sQ + h * HD_;
      const int n0 = lane, n1 = lane + 32;
      float s0 = dot16(qh, sK + n0 * KVPAD + h * HD_) * scale;
      float s1 = dot16(qh, sK + n1 * KVPAD + h * HD_) * scale;
      s0 = (sMsk[n0] != 0.f) ? s0 : -1e9f;
      s1 = (sMsk[n1] != 0.f) ? s1 : -1e9f;
      float m = wave_max(fmaxf(s0, s1));
      float e0 = __expf(s0 - m), e1 = __expf(s1 - m);
      float denom = wave_sum(e0 + e1);
      float inv = 1.f / denom;
      sAtt[h * N_ + n0] = e0 * inv;
      sAtt[h * N_ + n1] = e1 * inv;
    }
    __syncthreads();

    // ---- ctx = attn @ V ; has_agent flag ---------------------------------
    if (tid < 128) {
      const int h = tid >> 4, d = tid & 15;
      const float* att = sAtt + h * N_;
      const float* vcol = sV + h * HD_ + d;
      float acc = 0.f;
#pragma unroll 8
      for (int n = 0; n < N_; ++n) acc = fmaf(att[n], vcol[n * KVPAD], acc);
      sCtx[tid] = acc;
    }
    if (tid == 255) {
      float f = 0.f;
      for (int n = 0; n < N_; ++n) f = fmaxf(f, sMsk[n]);
      sRed[2] = f;
    }
    __syncthreads();

    // ---- out projection + residual ---------------------------------------
    if (tid < 128) {
      const float* wo = out_w + (size_t)tid * E_;
      float acc = out_b[tid];
#pragma unroll 8
      for (int j = 0; j < 128; ++j) acc = fmaf(sCtx[j], wo[j], acc);
      sEnc[tid] = sT[tid] + ((sRed[2] != 0.f) ? acc : 0.f);
    }
    __syncthreads();

    // ---- LayerNorm stats (wave 0) ----------------------------------------
    if (wv == 0) {
      float a = sEnc[lane], b = sEnc[lane + 32],
            c = sEnc[lane + 64], d = sEnc[lane + 96];
      float s  = wave_sum(a + b + c + d);
      float s2 = wave_sum(a * a + b * b + c * c + d * d);
      if (lane == 0) {
        float mu  = s * (1.f / 128.f);
        float var = s2 * (1.f / 128.f) - mu * mu;
        sRed[0] = mu;
        sRed[1] = rsqrtf(var + 1e-5f);
      }
    }
    __syncthreads();

    if (tid < 128) {
      float y = (sEnc[tid] - sRed[0]) * sRed[1];
      out[(size_t)row * E_ + tid] = y * ln_g[tid] + ln_b[tid];
    }
    __syncthreads();   // protect sA/sT/sMsk before next row's staging
  }
}

extern "C" void kernel_launch(void* const* d_in, const int* in_sizes, int n_in,
                              void* d_out, int out_size, void* d_ws, size_t ws_size,
                              hipStream_t stream) {
  const float*         target_emb = (const float*)d_in[0];
  const float*         agent_emb  = (const float*)d_in[1];
  const unsigned char* agent_mask = (const unsigned char*)d_in[2];
  const float*         in_proj_w  = (const float*)d_in[3];
  const float*         in_proj_b  = (const float*)d_in[4];
  const float*         out_w      = (const float*)d_in[5];
  const float*         out_b      = (const float*)d_in[6];
  const float*         ln_g       = (const float*)d_in[7];
  const float*         ln_b       = (const float*)d_in[8];
  float*               out        = (float*)d_out;

  dim3 grid(B_ / ROWS_PER_BLOCK);   // 2048 blocks, 4 rows each
  dim3 block(THREADS);              // 8 wave32 waves
  social_encoder_kernel<<<grid, block, SMEM_BYTES, stream>>>(
      target_emb, agent_emb, agent_mask, in_proj_w, in_proj_b,
      out_w, out_b, ln_g, ln_b, out);
}
